// PhysicsInformedKSpaceEncoder_41841571397637
// MI455X (gfx1250) — compile-verified, matrix operand order checked by reading, not run on hardware
//
#include <hip/hip_runtime.h>
#include <hip/hip_bf16.h>

// ---------------------------------------------------------------------------
// PhysicsInformedKSpaceEncoder forward for MI455X (gfx1250, wave32, WMMA).
// f16 WMMA (16x16x32) with f32 accumulation; weights pre-packed into WMMA
// B-fragment order so each lane loads its fragment as one 32B vector
// (2x global_load_b128). h stays L2-resident; segment-sum via f32 atomics.
// ---------------------------------------------------------------------------

#define HD 128          // hidden dim
#define EPSLN 1e-5f

typedef __attribute__((ext_vector_type(16))) _Float16 v16h;
typedef __attribute__((ext_vector_type(8)))  float    v8f;

// ---- WMMA fragment helpers (wave32, 16x16x32 f16 -> f32) -------------------
// 16-bit A matrix 16x32 layout (ISA 7.12.2): lanes 0-15 hold M=lane with
// K in {0..7,16..23}; lanes 16-31 hold M=lane-16 with K in {8..15,24..31}.
__device__ __forceinline__ int frag_k(int i, int hi) {
  return (i & 7) + ((i >> 3) << 4) + (hi << 3);
}

// A fragment from an LDS tile, row-major [16][ldk] halves, K-chunk at k0.
// Consecutive halves are contiguous -> compiler emits 2x ds_load_b128.
__device__ __forceinline__ v16h load_a_lds(const _Float16* t, int ldk, int k0) {
  int lane = threadIdx.x & 31;
  int m = lane & 15, hi = lane >> 4;
  v16h f;
#pragma unroll
  for (int i = 0; i < 16; ++i) f[i] = t[m * ldk + k0 + frag_k(i, hi)];
  return f;
}

__device__ __forceinline__ v8f wmma_f16(v16h a, v16h b, v8f c) {
  return __builtin_amdgcn_wmma_f32_16x16x32_f16(false, a, false, b,
                                                (short)0, c, false, false);
}

// Block-wide tree-reduce sum over nt threads (nt power of two), value to all.
__device__ __forceinline__ float block_reduce_sum(float v, float* red, int c, int nt) {
  red[c] = v;
  __syncthreads();
  for (int s = nt >> 1; s > 0; s >>= 1) {
    if (c < s) red[c] += red[c + s];
    __syncthreads();
  }
  float r = red[0];
  __syncthreads();
  return r;
}

// ---------------------------------------------------------------------------
// Weight packing: row-major f32 W[dk][dn] -> f16 fragments in B-fragment
// order: packed[((kc*numTiles + nt)*32 + lane)*16 + i], so one lane's
// fragment for chunk (kc, nt) is 32 contiguous bytes.
// ---------------------------------------------------------------------------
__global__ void pack_weight_kernel(const float* __restrict__ w,
                                   _Float16* __restrict__ packed,
                                   int dk, int dn) {
  int id = blockIdx.x * 256 + threadIdx.x;
  if (id < dk * dn) {
    int i = id & 15;
    int lane = (id >> 4) & 31;
    int rest = id >> 9;
    int numTiles = dn >> 4;
    int nt = rest % numTiles;
    int kc = rest / numTiles;
    int k = kc * 32 + frag_k(i, lane >> 4);
    int n = (nt << 4) + (lane & 15);
    packed[id] = (_Float16)w[k * dn + n];
  }
}

// ---------------------------------------------------------------------------
// utility kernels
// ---------------------------------------------------------------------------
__global__ void f32_to_f16_kernel(const float* __restrict__ s,
                                  _Float16* __restrict__ d, int n) {
  int i = blockIdx.x * 256 + threadIdx.x;
  if (i < n) d[i] = (_Float16)s[i];
}

__global__ void zero_kernel(float* __restrict__ p, size_t n) {
  size_t i = (size_t)blockIdx.x * 256 + threadIdx.x;
  if (i < n) p[i] = 0.0f;
}

// h = relu(inproj) + ke + sym_emb[label];  also f16 copy
__global__ void fuse_h0_kernel(const float* __restrict__ t0, const float* __restrict__ ke,
                               const float* __restrict__ emb, const int* __restrict__ lab,
                               float* __restrict__ h, _Float16* __restrict__ h16, int rows) {
  int i = blockIdx.x * 256 + threadIdx.x;
  if (i < rows * HD) {
    int row = i >> 7, c = i & 127;
    float v = t0[i] + ke[i] + emb[lab[row] * HD + c];
    h[i] = v;
    h16[i] = (_Float16)v;
  }
}

__global__ void concat16_kernel(const float* __restrict__ a, const float* __restrict__ b,
                                _Float16* __restrict__ o, int rows) {
  int i = blockIdx.x * 256 + threadIdx.x;
  if (i < rows * 256) {
    int row = i >> 8, c = i & 255;
    float v = (c < 128) ? a[row * 128 + c] : b[row * 128 + (c - 128)];
    o[i] = (_Float16)v;
  }
}

// ---------------------------------------------------------------------------
// Generic node GEMM (compile-time shapes so the K-loop fully unrolls into a
// wmma chain): out[rows,DN] = act(A16[rows,DK] @ Wp + bias)
// Block = 16 rows x 128 cols, 8 waves, one 16-col tile per wave.
// ---------------------------------------------------------------------------
template <int DK, int DN, bool RELU, bool OUT16>
__global__ __launch_bounds__(256) void node_gemm_kernel(
    const _Float16* __restrict__ A16, const v16h* __restrict__ Wp,
    const float* __restrict__ bias, void* __restrict__ outp, int rows) {
  constexpr int NT = DN / 16;   // n-tiles
  constexpr int CPR = DK / 8;   // uint4 chunks per A row
  __shared__ _Float16 sA[16 * DK];

  int tid = threadIdx.x;
  int r0 = blockIdx.x * 16;

  const uint4* Av = (const uint4*)A16;
  uint4* sAv = (uint4*)sA;
  for (int c = tid; c < 16 * CPR; c += 256) {
    int row = c / CPR, ch = c % CPR;
    int rr = min(r0 + row, rows - 1);
    sAv[row * CPR + ch] = Av[(size_t)rr * CPR + ch];
  }
  __syncthreads();

  int wave = tid >> 5;
  if (DN < 128 && (wave << 4) >= DN) return;
  int lane = tid & 31;

  v8f acc = {};
#pragma unroll
  for (int kc = 0; kc < DK / 32; ++kc) {
    v16h a = load_a_lds(sA, DK, kc * 32);
    v16h b = Wp[(kc * NT + wave) * 32 + lane];   // one 32B vector load
    acc = wmma_f16(a, b, acc);
  }
  int n = (wave << 4) + (lane & 15), hi = lane >> 4;
  float bv = bias[n];
#pragma unroll
  for (int r = 0; r < 8; ++r) {
    int row = r0 + r + (hi << 3);
    if (row < rows) {
      float v = acc[r] + bv;
      if (RELU) v = fmaxf(v, 0.0f);
      if (OUT16) ((_Float16*)outp)[row * DN + n] = (_Float16)v;
      else       ((float*)outp)[row * DN + n] = v;
    }
  }
}

// ---------------------------------------------------------------------------
// Fused edge-message layer. One block = 16 edges x 128 outputs.
//   m = W2 @ relu(LN(W1 @ [h_dst | h_src | k_diff] + b1)) + b2
//   agg[dst] += m   (fp32 global atomics)
// ---------------------------------------------------------------------------
__global__ __launch_bounds__(256) void edge_layer_kernel(
    const _Float16* __restrict__ h16, const float* __restrict__ kpos,
    const int* __restrict__ eidx, int nE,
    const v16h* __restrict__ w1p,          // packed first 256 K-rows of berry.l1
    const float* __restrict__ w1f,         // fp32 berry.l1 (for k_diff rows 256..258)
    const float* __restrict__ b1,
    const float* __restrict__ lng, const float* __restrict__ lnb,
    const v16h* __restrict__ w2p, const float* __restrict__ b2,
    float* __restrict__ agg) {
  __shared__ _Float16 sA[16 * 256];    // gathered [h_dst | h_src] tile (f16)
  __shared__ float    sM[16 * 128];    // GEMM1 out (pre-LN, f32)
  __shared__ _Float16 sM16[16 * 128];  // post LN/ReLU (f16)
  __shared__ float sKd[16 * 3];
  __shared__ int sSrc[16], sDst[16];
  __shared__ float sMu[16], sRs[16];

  int tid = threadIdx.x;
  int e0 = blockIdx.x * 16;

  if (tid < 16) {
    int e = min(e0 + tid, nE - 1);
    int s = eidx[e];        // edge_index[0] = src
    int d = eidx[nE + e];   // edge_index[1] = dst
    sSrc[tid] = s;
    sDst[tid] = d;
    sKd[tid * 3 + 0] = kpos[d * 3 + 0] - kpos[s * 3 + 0];
    sKd[tid * 3 + 1] = kpos[d * 3 + 1] - kpos[s * 3 + 1];
    sKd[tid * 3 + 2] = kpos[d * 3 + 2] - kpos[s * 3 + 2];
  }
  if (tid == 0) {   // hint next tile's indices into cache (global_prefetch_b8)
    __builtin_prefetch(&eidx[min(e0 + 16, nE - 1)], 0, 0);
    __builtin_prefetch(&eidx[nE + min(e0 + 16, nE - 1)], 0, 0);
  }
  __syncthreads();

  // Gather tile: 16 rows x 32 uint4 (dst 128 halves, then src 128 halves).
  const uint4* hv = (const uint4*)h16;   // 16 uint4 per node row
  uint4* sAv = (uint4*)sA;
  for (int c = tid; c < 512; c += 256) {
    int row = c >> 5, ch = c & 31;
    int node = (ch < 16) ? sDst[row] : sSrc[row];
    sAv[row * 32 + ch] = hv[(size_t)node * 16 + (ch & 15)];
  }
  __syncthreads();

  int wave = tid >> 5, lane = tid & 31;
  int nc = (wave << 4) + (lane & 15);
  int hi = lane >> 4;

  // GEMM1: K = 256 (h_dst | h_src), packed B fragments
  v8f acc = {};
#pragma unroll
  for (int kc = 0; kc < 8; ++kc) {
    v16h a = load_a_lds(sA, 256, kc * 32);
    v16h b = w1p[(kc * 8 + wave) * 32 + lane];
    acc = wmma_f16(a, b, acc);
  }
  // Epilogue: rank-3 k_diff rows (fp32 weights) + bias, stash pre-LN tile.
  float kw0 = w1f[256 * HD + nc], kw1 = w1f[257 * HD + nc], kw2 = w1f[258 * HD + nc];
  float bb = b1[nc];
#pragma unroll
  for (int r = 0; r < 8; ++r) {
    int m = r + (hi << 3);
    sM[m * HD + nc] = acc[r] + bb + sKd[m * 3 + 0] * kw0 +
                      sKd[m * 3 + 1] * kw1 + sKd[m * 3 + 2] * kw2;
  }
  __syncthreads();

  // Per-edge LayerNorm stats over 128 channels.
  if (tid < 16) {
    float s = 0.0f, ss = 0.0f;
    for (int k = 0; k < HD; ++k) {
      float t = sM[tid * HD + k];
      s += t; ss += t * t;
    }
    float mu = s * (1.0f / HD);
    sMu[tid] = mu;
    sRs[tid] = rsqrtf(ss * (1.0f / HD) - mu * mu + EPSLN);
  }
  __syncthreads();

  for (int i = tid; i < 16 * HD; i += 256) {
    int m = i >> 7, n = i & 127;
    float t = (sM[i] - sMu[m]) * sRs[m] * lng[n] + lnb[n];
    sM16[i] = (_Float16)fmaxf(t, 0.0f);
  }
  __syncthreads();

  // GEMM2: K = 128
  v8f acc2 = {};
#pragma unroll
  for (int kc = 0; kc < 4; ++kc) {
    v16h a = load_a_lds(sM16, HD, kc * 32);
    v16h b = w2p[(kc * 8 + wave) * 32 + lane];
    acc2 = wmma_f16(a, b, acc2);
  }
  float b2v = b2[nc];
#pragma unroll
  for (int r = 0; r < 8; ++r) {
    int m = r + (hi << 3);
    if (e0 + m < nE)
      unsafeAtomicAdd(&agg[sDst[m] * HD + nc], acc2[r] + b2v);
  }
}

// ---------------------------------------------------------------------------
// BZ positional encoding stage 1: relu(LN(k_pos @ W[3,128] + b)) -> f16
// ---------------------------------------------------------------------------
__global__ __launch_bounds__(128) void bz_l1_kernel(
    const float* __restrict__ kpos, const float* __restrict__ w,
    const float* __restrict__ b, const float* __restrict__ g,
    const float* __restrict__ lb, _Float16* __restrict__ out, int rows) {
  __shared__ float red[128];
  int row = blockIdx.x, c = threadIdx.x;
  float k0 = kpos[row * 3 + 0], k1 = kpos[row * 3 + 1], k2 = kpos[row * 3 + 2];
  float v = k0 * w[c] + k1 * w[HD + c] + k2 * w[2 * HD + c] + b[c];
  float mu = block_reduce_sum(v, red, c, 128) * (1.0f / HD);
  float d = v - mu;
  float var = block_reduce_sum(d * d, red, c, 128) * (1.0f / HD);
  out[row * HD + c] = (_Float16)fmaxf(d * rsqrtf(var + EPSLN) * g[c] + lb[c], 0.0f);
}

// ---------------------------------------------------------------------------
// Combine epilogue: o = relu(LN1(cpre)); h = LN2(o + h); refresh f16 copy.
// ---------------------------------------------------------------------------
__global__ __launch_bounds__(128) void combine_post_kernel(
    const float* __restrict__ cpre,
    const float* __restrict__ g1, const float* __restrict__ b1,
    const float* __restrict__ g2, const float* __restrict__ b2,
    float* __restrict__ h, _Float16* __restrict__ h16, int rows) {
  __shared__ float red[128];
  int row = blockIdx.x, c = threadIdx.x;
  int idx = row * HD + c;
  float v = cpre[idx];
  float mu = block_reduce_sum(v, red, c, 128) * (1.0f / HD);
  float d = v - mu;
  float var = block_reduce_sum(d * d, red, c, 128) * (1.0f / HD);
  float o = fmaxf(d * rsqrtf(var + EPSLN) * g1[c] + b1[c], 0.0f);
  float r = o + h[idx];
  float mu2 = block_reduce_sum(r, red, c, 128) * (1.0f / HD);
  float d2 = r - mu2;
  float var2 = block_reduce_sum(d2 * d2, red, c, 128) * (1.0f / HD);
  float hn = d2 * rsqrtf(var2 + EPSLN) * g2[c] + b2[c];
  h[idx] = hn;
  h16[idx] = (_Float16)hn;
}

// ---------------------------------------------------------------------------
// Global mean pool (atomic accumulate) and divide.
// ---------------------------------------------------------------------------
__global__ void pool_accum_kernel(const float* __restrict__ h, const int* __restrict__ batch,
                                  float* __restrict__ gsum, float* __restrict__ cnt, int rows) {
  int i = blockIdx.x * 256 + threadIdx.x;
  if (i < rows * HD) {
    int row = i >> 7, c = i & 127;
    int b = batch[row];
    unsafeAtomicAdd(&gsum[b * HD + c], h[i]);
    if (c == 0) unsafeAtomicAdd(&cnt[b], 1.0f);
  }
}

__global__ void pool_div_kernel(const float* __restrict__ gsum, const float* __restrict__ cnt,
                                float* __restrict__ g, int Bg) {
  int i = blockIdx.x * 256 + threadIdx.x;
  if (i < Bg * HD) g[i] = gsum[i] / fmaxf(cnt[i >> 7], 1.0f);
}

// ---------------------------------------------------------------------------
// Topological invariant head: 128 -> 64 (LN, relu) -> 32 (relu) -> dout.
// One block (64 threads) per graph.
// ---------------------------------------------------------------------------
__global__ __launch_bounds__(64) void topo_head_kernel(
    const float* __restrict__ g,
    const float* __restrict__ w1, const float* __restrict__ b1,
    const float* __restrict__ lng, const float* __restrict__ lnb,
    const float* __restrict__ w2, const float* __restrict__ b2,
    const float* __restrict__ w3, const float* __restrict__ b3,
    float* __restrict__ out, int dout, int stride, int off) {
  __shared__ float red[64];
  __shared__ float s1[64], s2[32];
  int gb = blockIdx.x, t = threadIdx.x;
  const float* gr = g + gb * HD;
  float a = b1[t];
  for (int k = 0; k < HD; ++k) a += gr[k] * w1[k * 64 + t];
  float mu = block_reduce_sum(a, red, t, 64) * (1.0f / 64.0f);
  float d = a - mu;
  float var = block_reduce_sum(d * d, red, t, 64) * (1.0f / 64.0f);
  s1[t] = fmaxf(d * rsqrtf(var + EPSLN) * lng[t] + lnb[t], 0.0f);
  __syncthreads();
  if (t < 32) {
    float a2 = b2[t];
    for (int k = 0; k < 64; ++k) a2 += s1[k] * w2[k * 32 + t];
    s2[t] = fmaxf(a2, 0.0f);
  }
  __syncthreads();
  if (t < dout) {
    float a3 = b3[t];
    for (int k = 0; k < 32; ++k) a3 += s2[k] * w3[k * dout + t];
    out[gb * stride + off + t] = a3;
  }
}

// Softmax over a length-1 axis is identically 1 -> att = (g@Wv+bv)@Wo+bo.
__global__ __launch_bounds__(128) void attn_kernel(
    const float* __restrict__ g, const float* __restrict__ wv,
    const float* __restrict__ bv, const float* __restrict__ wo,
    const float* __restrict__ bo, float* __restrict__ att) {
  __shared__ float sv[128];
  int gb = blockIdx.x, t = threadIdx.x;
  float a = bv[t];
  for (int k = 0; k < HD; ++k) a += g[gb * HD + k] * wv[k * HD + t];
  sv[t] = a;
  __syncthreads();
  float o = bo[t];
  for (int k = 0; k < HD; ++k) o += sv[k] * wo[k * HD + t];
  att[gb * HD + t] = o;
}

// Final: relu(LN([att | topo] @ W[135,256] + b))
__global__ __launch_bounds__(256) void out_kernel(
    const float* __restrict__ att, const float* __restrict__ topo,
    const float* __restrict__ w, const float* __restrict__ b,
    const float* __restrict__ g, const float* __restrict__ lb,
    float* __restrict__ out) {
  __shared__ float red[256];
  int gb = blockIdx.x, t = threadIdx.x;
  float a = b[t];
  for (int k = 0; k < HD; ++k) a += att[gb * HD + k] * w[k * 256 + t];
  for (int k = 0; k < 7; ++k) a += topo[gb * 7 + k] * w[(HD + k) * 256 + t];
  float mu = block_reduce_sum(a, red, t, 256) * (1.0f / 256.0f);
  float d = a - mu;
  float var = block_reduce_sum(d * d, red, t, 256) * (1.0f / 256.0f);
  out[gb * 256 + t] = fmaxf(d * rsqrtf(var + EPSLN) * g[t] + lb[t], 0.0f);
}

// ---------------------------------------------------------------------------
// Host driver.
// Input layout assumption: recursive insertion-order flattening of the
// setup_inputs() dict:
//   0:x 1:k_pos
//   2:ip.w 3:ip.b 4:sym_emb 5:bz.l1.w 6:bz.l1.b 7:bz.ln.g 8:bz.ln.b 9:bz.l2.w 10:bz.l2.b
//   layer l base = 11+16*l:
//     +0 berry.l1.w +1 berry.l1.b +2 berry.ln.g +3 berry.ln.b +4 berry.l2.w +5 berry.l2.b
//     +6 en.l1.w +7 en.l1.b +8 en.l2.w +9 en.l2.b
//     +10 comb.w +11 comb.b +12 comb.ln.g +13 comb.ln.b +14 post.g +15 post.b
//   75..98 topo (chern/z2/mirror x {l1w,l1b,lng,lnb,l2w,l2b,l3w,l3b})
//   99..106 attn q/k/v/o {w,b}   107..110 out {w,b,g,lb}
//   111 edge_index  112 batch  113 symmetry_labels
// ---------------------------------------------------------------------------
#define PF(i) ((const float*)d_in[(i)])

extern "C" void kernel_launch(void* const* d_in, const int* in_sizes, int n_in,
                              void* d_out, int out_size, void* d_ws, size_t ws_size,
                              hipStream_t stream) {
  (void)n_in; (void)ws_size;
  const float* x      = PF(0);
  const float* kpos   = PF(1);
  const float* ip_b   = PF(3);
  const float* symemb = PF(4);
  const int* eidx   = (const int*)d_in[111];
  const int* batch  = (const int*)d_in[112];
  const int* labels = (const int*)d_in[113];

  int Nn = in_sizes[112];       // nodes
  int Ee = in_sizes[111] / 2;   // edges
  int Bg = out_size / 256;      // graphs

  // ---- workspace carve (256B aligned) ----
  char* cur = (char*)d_ws;
  auto walloc = [&](size_t bytes) -> char* {
    char* p = cur;
    cur += (bytes + 255) & ~(size_t)255;
    return p;
  };
  float*    hF    = (float*)walloc((size_t)Nn * HD * 4);
  _Float16* hH    = (_Float16*)walloc((size_t)Nn * HD * 2);
  float*    aggF  = (float*)walloc((size_t)Nn * HD * 4);
  float*    eF    = (float*)walloc((size_t)Nn * HD * 4);
  float*    tmpF  = (float*)walloc((size_t)Nn * HD * 4);
  _Float16* act16 = (_Float16*)walloc((size_t)Nn * 256 * 2);
  _Float16* w_ip16 = (_Float16*)walloc(64 * HD * 2);
  _Float16* w_bz2p = (_Float16*)walloc(HD * HD * 2);
  _Float16 *wl1p[4], *wl2p[4], *we1p[4], *we2p[4], *wcp[4];
  for (int l = 0; l < 4; ++l) {
    wl1p[l] = (_Float16*)walloc(256 * HD * 2);   // first 256 K-rows, packed
    wl2p[l] = (_Float16*)walloc(HD * HD * 2);
    we1p[l] = (_Float16*)walloc(HD * 64 * 2);
    we2p[l] = (_Float16*)walloc(64 * HD * 2);
    wcp[l]  = (_Float16*)walloc(256 * HD * 2);
  }
  float* gsum    = (float*)walloc((size_t)Bg * HD * 4);
  float* cnt     = (float*)walloc((size_t)Bg * 4);
  float* gBuf    = (float*)walloc((size_t)Bg * HD * 4);
  float* topoBuf = (float*)walloc((size_t)Bg * 7 * 4);
  float* attBuf  = (float*)walloc((size_t)Bg * HD * 4);

  auto pack = [&](const float* s, _Float16* d, int dk, int dn) {
    pack_weight_kernel<<<(dk * dn + 255) / 256, 256, 0, stream>>>(s, d, dk, dn);
  };
  auto cvt = [&](const float* s, _Float16* d, int n) {
    f32_to_f16_kernel<<<(n + 255) / 256, 256, 0, stream>>>(s, d, n);
  };

  // ---- weight packing (f32 -> f16 fragment order) ----
  pack(PF(2), w_ip16, 64, HD);
  pack(PF(9), w_bz2p, HD, HD);
  for (int l = 0; l < 4; ++l) {
    int b = 11 + 16 * l;
    pack(PF(b + 0), wl1p[l], 256, HD);   // rows 256..258 handled in fp32 epilogue
    pack(PF(b + 4), wl2p[l], HD, HD);
    pack(PF(b + 6), we1p[l], HD, 64);
    pack(PF(b + 8), we2p[l], 64, HD);
    pack(PF(b + 10), wcp[l], 256, HD);
  }

  int nodeBlocks = (Nn + 15) / 16;

  // ---- input projection + BZ encoding ----
  cvt(x, act16, Nn * 64);
  node_gemm_kernel<64, 128, true, false><<<nodeBlocks, 256, 0, stream>>>(
      act16, (const v16h*)w_ip16, ip_b, tmpF, Nn);
  bz_l1_kernel<<<Nn, 128, 0, stream>>>(kpos, PF(5), PF(6), PF(7), PF(8), act16, Nn);
  node_gemm_kernel<128, 128, false, false><<<nodeBlocks, 256, 0, stream>>>(
      act16, (const v16h*)w_bz2p, PF(10), aggF, Nn);
  fuse_h0_kernel<<<(Nn * HD + 255) / 256, 256, 0, stream>>>(
      tmpF, aggF, symemb, labels, hF, hH, Nn);

  // ---- Berry curvature layers ----
  for (int l = 0; l < 4; ++l) {
    int b = 11 + 16 * l;
    zero_kernel<<<((size_t)Nn * HD + 255) / 256, 256, 0, stream>>>(aggF, (size_t)Nn * HD);
    edge_layer_kernel<<<(Ee + 15) / 16, 256, 0, stream>>>(
        hH, kpos, eidx, Ee,
        (const v16h*)wl1p[l], PF(b + 0), PF(b + 1), PF(b + 2), PF(b + 3),
        (const v16h*)wl2p[l], PF(b + 5), aggF);
    node_gemm_kernel<128, 64, true, true><<<nodeBlocks, 256, 0, stream>>>(
        hH, (const v16h*)we1p[l], PF(b + 7), act16, Nn);        // e1 -> f16 direct
    node_gemm_kernel<64, 128, false, false><<<nodeBlocks, 256, 0, stream>>>(
        act16, (const v16h*)we2p[l], PF(b + 9), eF, Nn);
    concat16_kernel<<<(Nn * 256 + 255) / 256, 256, 0, stream>>>(aggF, eF, act16, Nn);
    node_gemm_kernel<256, 128, false, false><<<nodeBlocks, 256, 0, stream>>>(
        act16, (const v16h*)wcp[l], PF(b + 11), tmpF, Nn);
    combine_post_kernel<<<Nn, 128, 0, stream>>>(
        tmpF, PF(b + 12), PF(b + 13), PF(b + 14), PF(b + 15), hF, hH, Nn);
  }

  // ---- pooling ----
  zero_kernel<<<((size_t)Bg * HD + 255) / 256, 256, 0, stream>>>(gsum, (size_t)Bg * HD);
  zero_kernel<<<1, 256, 0, stream>>>(cnt, (size_t)Bg);
  pool_accum_kernel<<<(Nn * HD + 255) / 256, 256, 0, stream>>>(hF, batch, gsum, cnt, Nn);
  pool_div_kernel<<<(Bg * HD + 255) / 256, 256, 0, stream>>>(gsum, cnt, gBuf, Bg);

  // ---- topo heads: chern(1) @0, z2(4) @1, mirror(2) @5 ----
  topo_head_kernel<<<Bg, 64, 0, stream>>>(gBuf, PF(75), PF(76), PF(77), PF(78),
                                          PF(79), PF(80), PF(81), PF(82), topoBuf, 1, 7, 0);
  topo_head_kernel<<<Bg, 64, 0, stream>>>(gBuf, PF(83), PF(84), PF(85), PF(86),
                                          PF(87), PF(88), PF(89), PF(90), topoBuf, 4, 7, 1);
  topo_head_kernel<<<Bg, 64, 0, stream>>>(gBuf, PF(91), PF(92), PF(93), PF(94),
                                          PF(95), PF(96), PF(97), PF(98), topoBuf, 2, 7, 5);

  // ---- attention (length-1 softmax == 1) + output head ----
  attn_kernel<<<Bg, 128, 0, stream>>>(gBuf, PF(103), PF(104), PF(105), PF(106), attBuf);
  out_kernel<<<Bg, 256, 0, stream>>>(attBuf, topoBuf, PF(107), PF(108), PF(109), PF(110),
                                     (float*)d_out);
}
